// MyNonlocal_48507360641756
// MI455X (gfx1250) — matrix operand order repaired and checked by reading, not compiled, not run on hardware
//
#include <hip/hip_runtime.h>

// ---------------------------------------------------------------------------
// Fused non-local block for MI455X (gfx1250), flash-style, bf16 WMMA,
// with async global->LDS (ASYNCcnt) double-buffered K/V tile staging.
// ---------------------------------------------------------------------------

#define N_    8
#define C_    64
#define CRED  32
#define H_    56
#define W_    56
#define HW    (H_ * W_)      // 3136
#define TILES (HW / 16)      // 196 row tiles
#define EPS_  1e-5f

typedef __bf16 bf16_t;
typedef __attribute__((ext_vector_type(16))) __bf16 v16bf;
typedef __attribute__((ext_vector_type(8)))  __bf16 v8bf;
typedef __attribute__((ext_vector_type(8)))  float  v8f;

// A-matrix fragment (16-bit, 16x32): lane half 0 -> K {0..7, 16..23},
// lane half 1 -> K {8..15, 24..31}.  rowptr points at 32 contiguous K.
__device__ __forceinline__ v16bf load_a32(const bf16_t* rowptr, int half) {
  v8bf lo = *(const v8bf*)(rowptr + half * 8);
  v8bf hi = *(const v8bf*)(rowptr + 16 + half * 8);
  v16bf r;
#pragma unroll
  for (int j = 0; j < 8; ++j) { r[j] = lo[j]; r[8 + j] = hi[j]; }
  return r;
}

// B-matrix fragment (16-bit, 32x16): lane half 0 -> K {0..7, 16..23},
// lane half 1 -> K {8..15, 24..31}.  rowptr points at 32 contiguous K
// for this lane's fixed N column.
__device__ __forceinline__ v16bf load_b32(const bf16_t* rowptr, int half) {
  v8bf lo = *(const v8bf*)(rowptr + half * 8);
  v8bf hi = *(const v8bf*)(rowptr + 16 + half * 8);
  v16bf r;
#pragma unroll
  for (int j = 0; j < 8; ++j) { r[j] = lo[j]; r[8 + j] = hi[j]; }
  return r;
}

__device__ __forceinline__ v8f wmma_bf16(v16bf a, v16bf b, v8f c) {
  return __builtin_amdgcn_wmma_f32_16x16x32_bf16(
      /*neg_a=*/false, a, /*neg_b=*/false, b,
      /*c_mod=*/(short)0, c, /*reuse_a=*/false, /*reuse_b=*/false);
}

// one async 512B global->LDS copy: 32 lanes x 16B (ASYNCcnt tracked)
__device__ __forceinline__ void async_copy_512(unsigned lds_addr,
                                               unsigned long long gaddr) {
  asm volatile("global_load_async_to_lds_b128 %0, %1, off"
               :: "v"(lds_addr), "v"(gaddr) : "memory");
}

// ---------------------------------------------------------------------------
// K0a: theta transpose -> q[n][ab][k] (bf16, row = 32 contiguous channels)
// ---------------------------------------------------------------------------
__global__ void k_prep_q(const float* __restrict__ l, bf16_t* __restrict__ q) {
  int idx = blockIdx.x * blockDim.x + threadIdx.x;   // n*HW + ab
  if (idx >= N_ * HW) return;
  int n = idx / HW, ab = idx % HW;
  const float* src = l + (size_t)n * C_ * HW + ab;
  bf16_t* dst = q + (size_t)idx * CRED;
#pragma unroll
  for (int k = 0; k < CRED; ++k) dst[k] = (bf16_t)src[(size_t)k * HW];
}

// ---------------------------------------------------------------------------
// K0b: g = relu6(BN1(w_inside (x) l)) stored as bf16 vt[n][i][cd]
// ---------------------------------------------------------------------------
#define GTILE 128
__global__ __launch_bounds__(GTILE) void k_conv1x1(
    const float* __restrict__ l, const float* __restrict__ w,
    const float* __restrict__ g1, const float* __restrict__ b1,
    const float* __restrict__ m1, const float* __restrict__ v1,
    bf16_t* __restrict__ vt) {
  __shared__ float lbuf[C_][GTILE];
  __shared__ float wbuf[C_ * C_];
  int n = blockIdx.y;
  int abBase = blockIdx.x * GTILE;
  int t = threadIdx.x;
  for (int idx = t; idx < C_ * C_; idx += GTILE) wbuf[idx] = w[idx];
  int ab = abBase + t;
  for (int k = 0; k < C_; ++k)
    lbuf[k][t] = (ab < HW) ? l[((size_t)n * C_ + k) * HW + ab] : 0.0f;
  __syncthreads();
  if (ab >= HW) return;
  for (int i = 0; i < C_; ++i) {
    float acc = 0.0f;
#pragma unroll 8
    for (int k = 0; k < C_; ++k) acc += wbuf[i * C_ + k] * lbuf[k][t];
    float inv = rsqrtf(v1[i] + EPS_);
    float val = (acc - m1[i]) * (inv * g1[i]) + b1[i];
    val = fminf(fmaxf(val, 0.0f), 6.0f);
    vt[((size_t)n * C_ + i) * HW + ab] = (bf16_t)val;
  }
}

// ---------------------------------------------------------------------------
// K1: column stats of s = theta^T theta via symmetry (col stats == row stats).
// One wave per 16-row tile, streaming (max, sum) over 196 WMMA score tiles.
// ---------------------------------------------------------------------------
__global__ __launch_bounds__(256) void k_colstats(
    const bf16_t* __restrict__ q, float* __restrict__ colmax,
    float* __restrict__ colrcp) {
  int wave = threadIdx.x >> 5, lane = threadIdx.x & 31;
  int task = blockIdx.x * 8 + wave;           // uniform per wave
  if (task >= N_ * TILES) return;
  int n = task / TILES, tile = task % TILES;
  int half = lane >> 4, lid = lane & 15;
  const bf16_t* qn = q + (size_t)n * HW * CRED;

  v16bf aQ = load_a32(qn + (size_t)(tile * 16 + lid) * CRED, half);

  float m[8], s[8];
#pragma unroll
  for (int r = 0; r < 8; ++r) { m[r] = -3.0e38f; s[r] = 0.0f; }

  for (int ab0 = 0; ab0 < HW; ab0 += 16) {
    v16bf bK = load_b32(qn + (size_t)(ab0 + lid) * CRED, half);
    v8f sc = {};
    sc = wmma_bf16(aQ, bK, sc);
#pragma unroll
    for (int r = 0; r < 8; ++r) {
      float x = sc[r];
      float mn = fmaxf(m[r], x);
      s[r] = s[r] * __expf(m[r] - mn) + __expf(x - mn);
      m[r] = mn;
    }
  }
  // merge (max, sum) across the 16 lanes holding one row's columns
#pragma unroll
  for (int r = 0; r < 8; ++r) {
    float mm = m[r], ss = s[r];
    for (int off = 1; off < 16; off <<= 1) {
      float om = __shfl_xor(mm, off, 16);
      float os = __shfl_xor(ss, off, 16);
      float mn = fmaxf(mm, om);
      ss = ss * __expf(mm - mn) + os * __expf(om - mn);
      mm = mn;
    }
    if (lid == 0) {
      int cd = tile * 16 + r + 8 * half;
      colmax[n * HW + cd] = mm;
      colrcp[n * HW + cd] = 1.0f / ss;
    }
  }
}

// ---------------------------------------------------------------------------
// K2: fused  out_pre = softmax_row(exp-col-norm(theta^T theta)) @ g^T
// One block per (batch, group of 8 ab tiles); wave 0 async-stages the shared
// K (32x32) and V (64x32) tiles into double-buffered LDS; each wave runs the
// score WMMA + double-exp + P@V WMMA accumulate out of LDS.
// ---------------------------------------------------------------------------
__global__ __launch_bounds__(256) void k_attn(
    const bf16_t* __restrict__ q, const bf16_t* __restrict__ vt,
    const float* __restrict__ colmax, const float* __restrict__ colrcp,
    float* __restrict__ outp) {
  __shared__ bf16_t sK[2][32 * 32];            // 2KB per buffer
  __shared__ bf16_t sV[2][64 * 32];            // 4KB per buffer
  __shared__ bf16_t ldsP[8][16 * 32];          // per-wave P staging tile

  int wave = threadIdx.x >> 5, lane = threadIdx.x & 31;
  int n = blockIdx.y;
  int tile = blockIdx.x * 8 + wave;            // uniform per wave
  bool active = (tile < TILES);
  int half = lane >> 4, lid = lane & 15;
  const bf16_t* qn = q + (size_t)n * HW * CRED;
  const bf16_t* vn = vt + (size_t)n * C_ * HW;
  const float* cmx = colmax + n * HW;
  const float* crc = colrcp + n * HW;
  bf16_t* P = ldsP[wave];

  const unsigned ldsKb = (unsigned)(size_t)&sK[0][0];
  const unsigned ldsVb = (unsigned)(size_t)&sV[0][0];

  // ---- async stage of one K/V tile pair into buffer `buf` for window cdN ---
  auto stage_tiles = [&](int buf, int cdN) {
    // K tile: q rows cdN..cdN+31 are contiguous (32 bf16/row) -> 2KB linear
    unsigned long long ks = (unsigned long long)(qn + (size_t)cdN * CRED);
    unsigned kl = ldsKb + (unsigned)(buf * 2048);
#pragma unroll
    for (int s = 0; s < 4; ++s)
      async_copy_512(kl + s * 512 + lane * 16,
                     ks + (unsigned)(s * 512 + lane * 16));
    // V tile: 64 channel-segments of 64B (stride HW*2B) -> channel-major LDS
#pragma unroll
    for (int s = 0; s < 8; ++s) {
      int seg = s * 32 + lane;                 // 16B chunk id, 0..255
      int ch = seg >> 2, sub = seg & 3;
      unsigned laddr = ldsVb + (unsigned)(buf * 4096 + ch * 64 + sub * 16);
      unsigned long long ga =
          (unsigned long long)(vn + (size_t)ch * HW + cdN) +
          (unsigned)(sub * 16);
      async_copy_512(laddr, ga);
    }
  };

  // preload buffer 0
  if (wave == 0) {
    stage_tiles(0, 0);
    asm volatile("s_wait_asynccnt 0" ::: "memory");
  }
  __syncthreads();

  v16bf aQ = {};
  v8f O[4];
#pragma unroll
  for (int ic = 0; ic < 4; ++ic) O[ic] = (v8f){};
  v8f den = {};
  if (active)
    aQ = load_a32(qn + (size_t)(tile * 16 + lid) * CRED, half);

  int it = 0;
  for (int cd0 = 0; cd0 < HW; cd0 += 32, ++it) {
    int cur = it & 1, nxt = cur ^ 1;
    if (wave == 0 && cd0 + 32 < HW)
      stage_tiles(nxt, cd0 + 32);              // overlap copy with compute

    if (active) {
      const bf16_t* sKc = &sK[cur][0];
      const bf16_t* sVc = &sV[cur][0];
#pragma unroll
      for (int c = 0; c < 2; ++c) {
        int cb = cd0 + 16 * c;
        v16bf bK = load_b32(sKc + (size_t)(16 * c + lid) * CRED, half);
        v8f sc = {};
        sc = wmma_bf16(aQ, bK, sc);
        float cm = cmx[cb + lid];
        float cr = crc[cb + lid];
#pragma unroll
        for (int r = 0; r < 8; ++r) {
          // f1 = exp(s - colmax)/colsum in (0,1];  p = exp(f1) in [1,e]
          float p = __expf(__expf(sc[r] - cm) * cr);
          den[r] += p;
          int mrow = r + 8 * half;
          P[mrow * 32 + 16 * c + lid] = (bf16_t)p;
        }
      }
      asm volatile("s_wait_dscnt 0" ::: "memory");   // cross-lane LDS RAW
      v16bf aP = load_a32(P + lid * 32, half);
#pragma unroll
      for (int ic = 0; ic < 4; ++ic) {
        v16bf bV = load_b32(sVc + (size_t)(ic * 16 + lid) * 32, half);
        O[ic] = wmma_bf16(aP, bV, O[ic]);
      }
    }

    if (wave == 0)
      asm volatile("s_wait_asynccnt 0" ::: "memory");
    __syncthreads();                           // publish buffer `nxt`
  }

  if (!active) return;

  // row sums of p across the 16 lanes of each half
  float rd[8];
#pragma unroll
  for (int r = 0; r < 8; ++r) {
    float d = den[r];
    for (int off = 1; off < 16; off <<= 1) d += __shfl_xor(d, off, 16);
    rd[r] = 1.0f / d;
  }

  const float sHW = (1.0f / (float)H_) * (float)W_;  // reference's /h*w
#pragma unroll
  for (int ic = 0; ic < 4; ++ic) {
    int i = ic * 16 + lid;
#pragma unroll
    for (int r = 0; r < 8; ++r) {
      int ab = tile * 16 + r + 8 * half;
      outp[((size_t)n * C_ + i) * HW + ab] = O[ic][r] * rd[r] * sHW;
    }
  }
}

// ---------------------------------------------------------------------------
// K3: depthwise 3x3 (pad 1) + BN2 + residual
// ---------------------------------------------------------------------------
__global__ void k_dwconv(
    const float* __restrict__ outp, const float* __restrict__ l,
    const float* __restrict__ wdw, const float* __restrict__ g2,
    const float* __restrict__ b2, const float* __restrict__ m2,
    const float* __restrict__ v2, float* __restrict__ out) {
  int idx = blockIdx.x * blockDim.x + threadIdx.x;
  if (idx >= N_ * C_ * HW) return;
  int w = idx % W_;
  int h = (idx / W_) % H_;
  int i = (idx / HW) % C_;
  const float* plane = outp + (idx - (h * W_ + w));
  float acc = 0.0f;
#pragma unroll
  for (int dy = -1; dy <= 1; ++dy)
#pragma unroll
    for (int dx = -1; dx <= 1; ++dx) {
      int hh = h + dy, ww = w + dx;
      if (hh >= 0 && hh < H_ && ww >= 0 && ww < W_)
        acc += wdw[i * 9 + (dy + 1) * 3 + (dx + 1)] * plane[hh * W_ + ww];
    }
  float inv = rsqrtf(v2[i] + EPS_);
  out[idx] = (acc - m2[i]) * (inv * g2[i]) + b2[i] + l[idx];
}

// ---------------------------------------------------------------------------
extern "C" void kernel_launch(void* const* d_in, const int* in_sizes, int n_in,
                              void* d_out, int out_size, void* d_ws,
                              size_t ws_size, hipStream_t stream) {
  (void)in_sizes; (void)n_in; (void)out_size; (void)ws_size;
  const float* l        = (const float*)d_in[0];
  const float* w_inside = (const float*)d_in[1];
  const float* gamma1   = (const float*)d_in[2];
  const float* beta1    = (const float*)d_in[3];
  const float* mean1    = (const float*)d_in[4];
  const float* var1     = (const float*)d_in[5];
  const float* w_dw     = (const float*)d_in[6];
  const float* gamma2   = (const float*)d_in[7];
  const float* beta2    = (const float*)d_in[8];
  const float* mean2    = (const float*)d_in[9];
  const float* var2     = (const float*)d_in[10];
  float* out = (float*)d_out;

  // workspace carve-up (~11.5 MB total)
  bf16_t* q      = (bf16_t*)d_ws;                 // N*HW*CRED bf16
  bf16_t* vt     = q + (size_t)N_ * HW * CRED;    // N*C*HW  bf16
  float*  colmax = (float*)(vt + (size_t)N_ * C_ * HW);
  float*  colrcp = colmax + (size_t)N_ * HW;
  float*  outp   = colrcp + (size_t)N_ * HW;      // N*C*HW fp32

  k_prep_q<<<(N_ * HW + 255) / 256, 256, 0, stream>>>(l, q);
  k_conv1x1<<<dim3((HW + GTILE - 1) / GTILE, N_), GTILE, 0, stream>>>(
      l, w_inside, gamma1, beta1, mean1, var1, vt);
  k_colstats<<<(N_ * TILES + 7) / 8, 256, 0, stream>>>(q, colmax, colrcp);
  k_attn<<<dim3((TILES + 7) / 8, N_), 256, 0, stream>>>(
      q, vt, colmax, colrcp, outp);
  k_dwconv<<<(N_ * C_ * HW + 255) / 256, 256, 0, stream>>>(
      outp, l, w_dw, gamma2, beta2, mean2, var2, out);
}
